// RNN_80264348828236
// MI455X (gfx1250) — matrix operand order repaired and checked by reading, not compiled, not run on hardware
//
#include <hip/hip_runtime.h>
#include <hip/hip_bf16.h>

// ---------------------------------------------------------------------------
// Stacked Elman RNN LM for gfx1250 (MI455X), wave32 + v_wmma_f32_16x16x32_f16.
// S=128, B=64, V=10000, E=H=512, L=2.
// ---------------------------------------------------------------------------

typedef __attribute__((ext_vector_type(16))) _Float16 v16h;
typedef __attribute__((ext_vector_type(8)))  _Float16 h8;
typedef __attribute__((ext_vector_type(8)))  float    v8f;

#define SEQ   128
#define BATCH 64
#define HID   512
#define VOCAB 10000
#define LAYERS 2

__device__ __forceinline__ float ftanh(float x) {
    float e = __expf(2.0f * x);
    return (e - 1.0f) / (e + 1.0f);
}

// Opaque uniform offset: regenerated by asm volatile each loop iteration, so
// weight-fragment loads can't be hoisted out of the sequential scan (which
// previously spilled 128 VGPRs). Laundering only the OFFSET (not the pointer)
// preserves the base pointer's global address-space provenance -> global_load.
__device__ __forceinline__ size_t opaque(size_t x) {
    asm volatile("" : "+s"(x));
    return x;
}

// A fragment (16x32 f16, tile row base mr, k base kb) from row-major [*,ld] f16.
// Per ISA 7.12.2: lanes 0-15 hold M=lane, K = kb+{0..7,16..23}; lanes 16-31: K = kb+{8..15,24..31}.
__device__ __forceinline__ v16h load_a(const _Float16* __restrict__ X, int ld,
                                       int mr, int kb, int lane) {
    int m = lane & 15, kh = lane >> 4;
    const _Float16* p = X + (size_t)(mr + m) * ld + kb + kh * 8;
    h8 lo = *(const h8*)(p);
    h8 hi = *(const h8*)(p + 16);
    return __builtin_shufflevector(lo, hi, 0,1,2,3,4,5,6,7,8,9,10,11,12,13,14,15);
}

// B fragment (32x16 f16, col base nb, k base kb) from K-major (transposed) weights Wt[N][K].
// Per ISA: lane n (0-15) holds column n, K = kb+{0..7,16..23} ; lanes 16-31: K = kb+{8..15,24..31}.
__device__ __forceinline__ v16h load_b(const _Float16* __restrict__ Wt, int ldk,
                                       int nb, int kb, int lane) {
    int n = lane & 15, kh = lane >> 4;
    const _Float16* p = Wt + (size_t)(nb + n) * ldk + kb + kh * 8;
    h8 lo = *(const h8*)(p);
    h8 hi = *(const h8*)(p + 16);
    return __builtin_shufflevector(lo, hi, 0,1,2,3,4,5,6,7,8,9,10,11,12,13,14,15);
}

// NT M-tiles (stride mstep rows) x one 16-wide N tile; B fragment reused across
// M-tiles. All NT+1 fragment loads of a k-slice are issued before any WMMA so
// the memory latency of a slice is overlapped instead of paid per-WMMA.
template <int NT>
__device__ __forceinline__ void gemm_mtiles(const _Float16* __restrict__ A, int lda,
                                            int mr0, int mstep,
                                            const _Float16* __restrict__ Bt, int ldb,
                                            int nb, int K, int lane, v8f acc[NT]) {
    for (int kb = 0; kb < K; kb += 32) {
        v16h b = load_b(Bt, ldb, nb, kb, lane);
        v16h a[NT];
#pragma unroll
        for (int i = 0; i < NT; ++i)
            a[i] = load_a(A, lda, mr0 + i * mstep, kb, lane);
#pragma unroll
        for (int i = 0; i < NT; ++i)
            acc[i] = __builtin_amdgcn_wmma_f32_16x16x32_f16(
                false, a[i], false, b, (short)0, acc[i], false, false);
    }
}

// ---- prep: f32 [K][N] row-major -> f16 [N][K] (K-contiguous for B fragments) ----
__global__ void transpose_f32_to_f16(const float* __restrict__ src,
                                     _Float16* __restrict__ dst, int K, int N) {
    __shared__ float tile[16][17];
    int tx = threadIdx.x & 15, ty = threadIdx.x >> 4;   // 256 threads
    int kb = blockIdx.y * 16, nbase = blockIdx.x * 16;
    int k = kb + ty, n = nbase + tx;
    if (k < K && n < N) tile[ty][tx] = src[(size_t)k * N + n];
    __syncthreads();
    int nn = nbase + ty, kk = kb + tx;
    if (nn < N && kk < K) dst[(size_t)nn * K + kk] = (_Float16)tile[tx][ty];
}

// ---- prep: embedding gather, f32 table -> f16 activations [S*B][E] ----
__global__ void embed_gather(const int* __restrict__ tokens,
                             const float* __restrict__ table,
                             _Float16* __restrict__ out) {
    int row = blockIdx.x;                 // s*BATCH + b
    int tok = tokens[row];
    for (int e = threadIdx.x; e < HID; e += blockDim.x)
        out[(size_t)row * HID + e] = (_Float16)table[(size_t)tok * HID + e];
}

// ---- persistent recurrence: one workgroup (32 wave32s), activations in 192KB LDS ----
__global__ void __launch_bounds__(1024, 1)
rnn_scan(const _Float16* __restrict__ emb16, const float* __restrict__ h_init,
         const _Float16* __restrict__ WinT, const float* __restrict__ b_in,
         const _Float16* __restrict__ WfcT, const float* __restrict__ b_fc,
         const _Float16* __restrict__ WrecT, const float* __restrict__ b_rec,
         _Float16* __restrict__ h_all, float* __restrict__ h_final) {
    extern __shared__ _Float16 smem[];
    _Float16* sInp = smem;                          // [64][512]
    _Float16* sH0  = smem + BATCH * HID;            // layer 0 hidden [64][512]
    _Float16* sH1  = smem + 2 * BATCH * HID;        // layer 1 hidden [64][512]

    const int tid  = threadIdx.x;
    const int wave = tid >> 5, lane = tid & 31;
    const int n0 = lane & 15, kh = lane >> 4;
    const int nb = wave * 16;                    // each wave owns one 16-wide N slab

    // load initial hidden state (f32 -> f16) into LDS; sH0|sH1 are contiguous
    for (int i = tid; i < LAYERS * BATCH * HID; i += 1024)
        smem[BATCH * HID + i] = (_Float16)h_init[i];
    __syncthreads();

    for (int t = 0; t < SEQ; ++t) {
        const _Float16* xt = emb16 + (size_t)t * (BATCH * HID);

        // Stage A: inp = tanh(x_t @ W_in + b_in)  -> LDS
        {
            const _Float16* wB = WinT + opaque(0);   // re-opaque per t: no cross-t hoist
            v8f acc[4]; v8f z = {0,0,0,0,0,0,0,0};
#pragma unroll
            for (int i = 0; i < 4; ++i) acc[i] = z;
            gemm_mtiles<4>(xt, HID, 0, 16, wB, HID, nb, HID, lane, acc);
            float bv = b_in[nb + n0];
#pragma unroll
            for (int i = 0; i < 4; ++i)
#pragma unroll
                for (int r = 0; r < 8; ++r) {
                    int row = i * 16 + r + kh * 8;
                    sInp[(size_t)row * HID + nb + n0] =
                        (_Float16)ftanh(acc[i][r] + bv);
                }
        }
        __syncthreads();

        // Layers: h_l = tanh(aIn @ W_fc[l] + h_l @ W_rec[l] + b_fc[l] + b_rec[l])
        const _Float16* aIn = sInp;
        for (int l = 0; l < LAYERS; ++l) {
            _Float16* hl = (l == 0) ? sH0 : sH1;   // runtime select, no LDS ptr aggregate
            const _Float16* wFc  = WfcT  + opaque((size_t)l * HID * HID);
            const _Float16* wRec = WrecT + opaque((size_t)l * HID * HID);
            v8f acc[4]; v8f z = {0,0,0,0,0,0,0,0};
#pragma unroll
            for (int i = 0; i < 4; ++i) acc[i] = z;
            gemm_mtiles<4>(aIn, HID, 0, 16, wFc,  HID, nb, HID, lane, acc);
            gemm_mtiles<4>(hl,  HID, 0, 16, wRec, HID, nb, HID, lane, acc);
            __syncthreads();   // all reads of hl/aIn done before overwrite
            float bv = b_fc[l * HID + nb + n0] + b_rec[l * HID + nb + n0];
#pragma unroll
            for (int i = 0; i < 4; ++i)
#pragma unroll
                for (int r = 0; r < 8; ++r) {
                    int row = i * 16 + r + kh * 8;
                    float v = ftanh(acc[i][r] + bv);
                    hl[(size_t)row * HID + nb + n0] = (_Float16)v;
                    if (l == LAYERS - 1)
                        h_all[((size_t)t * BATCH + row) * HID + nb + n0] = (_Float16)v;
                    if (t == SEQ - 1)
                        h_final[((size_t)l * BATCH + row) * HID + nb + n0] = v;
                }
            __syncthreads();
            aIn = hl;
        }
    }
}

// ---- output projection: [S*B,512] x [512,10000] + b_out, chip-wide WMMA GEMM ----
__global__ void __launch_bounds__(256)
logits_gemm(const _Float16* __restrict__ h_all, const _Float16* __restrict__ WoutT,
            const float* __restrict__ b_out, float* __restrict__ out) {
    const int wave = threadIdx.x >> 5, lane = threadIdx.x & 31;
    const int n0 = lane & 15, kh = lane >> 4;
    const int nb  = blockIdx.x * 16;                     // 625 N tiles (10000 = 625*16)
    const int mr0 = (blockIdx.y * 8 + wave) * 64;        // 4 M-tiles of 16 rows per wave

    v8f acc[4]; v8f z = {0,0,0,0,0,0,0,0};
#pragma unroll
    for (int i = 0; i < 4; ++i) acc[i] = z;
    gemm_mtiles<4>(h_all, HID, mr0, 16, WoutT, HID, nb, HID, lane, acc);

    float bv = b_out[nb + n0];
#pragma unroll
    for (int i = 0; i < 4; ++i)
#pragma unroll
        for (int r = 0; r < 8; ++r) {
            int row = mr0 + i * 16 + r + kh * 8;
            out[(size_t)row * VOCAB + nb + n0] = acc[i][r] + bv;
        }
}

extern "C" void kernel_launch(void* const* d_in, const int* in_sizes, int n_in,
                              void* d_out, int out_size, void* d_ws, size_t ws_size,
                              hipStream_t stream) {
    const int*   inputs    = (const int*)d_in[0];
    const float* hidden    = (const float*)d_in[1];
    const float* emb_table = (const float*)d_in[2];
    const float* W_in      = (const float*)d_in[3];
    const float* b_in      = (const float*)d_in[4];
    const float* W_fc      = (const float*)d_in[5];
    const float* b_fc      = (const float*)d_in[6];
    const float* W_rec     = (const float*)d_in[7];
    const float* b_rec     = (const float*)d_in[8];
    const float* W_out     = (const float*)d_in[9];
    const float* b_out     = (const float*)d_in[10];

    float* logits  = (float*)d_out;
    float* h_final = logits + (size_t)SEQ * BATCH * VOCAB;

    // workspace layout (f16 elements)
    _Float16* ws     = (_Float16*)d_ws;
    _Float16* emb16  = ws;                                       // S*B*H
    _Float16* WinT   = emb16 + (size_t)SEQ * BATCH * HID;        // H*H
    _Float16* WfcT   = WinT  + (size_t)HID * HID;                // 2*H*H
    _Float16* WrecT  = WfcT  + (size_t)LAYERS * HID * HID;       // 2*H*H
    _Float16* WoutT  = WrecT + (size_t)LAYERS * HID * HID;       // V*H
    _Float16* h_all  = WoutT + (size_t)VOCAB * HID;              // S*B*H

    // weight prep: f32 [K][N] -> f16 [N][K]
    dim3 tg512(HID / 16, HID / 16);
    transpose_f32_to_f16<<<tg512, 256, 0, stream>>>(W_in, WinT, HID, HID);
    for (int l = 0; l < LAYERS; ++l) {
        transpose_f32_to_f16<<<tg512, 256, 0, stream>>>(
            W_fc + (size_t)l * HID * HID, WfcT + (size_t)l * HID * HID, HID, HID);
        transpose_f32_to_f16<<<tg512, 256, 0, stream>>>(
            W_rec + (size_t)l * HID * HID, WrecT + (size_t)l * HID * HID, HID, HID);
    }
    dim3 tgout(VOCAB / 16, HID / 16);
    transpose_f32_to_f16<<<tgout, 256, 0, stream>>>(W_out, WoutT, HID, VOCAB);

    // embedding gather -> f16
    embed_gather<<<SEQ * BATCH, 256, 0, stream>>>(inputs, emb_table, emb16);

    // persistent recurrence on one WGP, activations in 192KB LDS
    size_t lds_bytes = (size_t)3 * BATCH * HID * sizeof(_Float16);  // 192 KB
    rnn_scan<<<1, 1024, lds_bytes, stream>>>(emb16, hidden, WinT, b_in,
                                             WfcT, b_fc, WrecT, b_rec,
                                             h_all, h_final);

    // hoisted output projection (84 GFLOP) across the whole chip
    logits_gemm<<<dim3(VOCAB / 16, 16), 256, 0, stream>>>(h_all, WoutT, b_out, logits);
}
